// Nmp1_38998303048178
// MI455X (gfx1250) — compile-verified, matrix operand order checked by reading, not run on hardware
//
#include <hip/hip_runtime.h>
#include <hip/hip_fp16.h>
#include <math.h>

typedef __attribute__((ext_vector_type(16))) _Float16 v16h;
typedef __attribute__((ext_vector_type(8)))  _Float16 v8h;
typedef __attribute__((ext_vector_type(8)))  float    v8f;

#define WMMA16(a, b, c) __builtin_amdgcn_wmma_f32_16x16x32_f16( \
    false, (a), false, (b), (short)0, (c), false, false)

// Load a 16x32 (f16) WMMA operand fragment from an LDS matrix.
// For A: `row` is the lane's matrix row (stored row-major, K contiguous).
// For B: `row` is the lane's matrix column (stored N-major, K contiguous).
// Per ISA 7.12.2: lanes 0-15 hold K {0..7,16..23}, lanes 16-31 hold K {8..15,24..31}.
__device__ inline v16h load_frag(const _Float16* smem, int row, int stride,
                                 int kc, int lane) {
  int hp = (lane >> 4) & 1;
  const _Float16* p = smem + row * stride + kc + hp * 8;
  v8h lo = *(const v8h*)(p);        // K = kc + hp*8 + 0..7
  v8h hi = *(const v8h*)(p + 16);   // K = kc + 16 + hp*8 + 0..7
  return __builtin_shufflevector(lo, hi, 0, 1, 2, 3, 4, 5, 6, 7,
                                 8, 9, 10, 11, 12, 13, 14, 15);
}

// ---------------------------------------------------------------------------
// Kernel 1: me[b,v,:] = sum_w g[b,v,w] * e[b,v,w,:], deg[b,v] = sum_w g[b,v,w]
// One wave per (b,v) row. Streams e (512 MB) exactly once, fully coalesced.
// ---------------------------------------------------------------------------
__global__ __launch_bounds__(256) void me_deg_kernel(const float* __restrict__ g,
                                                     const float* __restrict__ e,
                                                     float* __restrict__ me,
                                                     float* __restrict__ deg) {
  __shared__ float sG[8][512];
  int lane = threadIdx.x & 31;
  int wv = threadIdx.x >> 5;
  int row = blockIdx.x * 8 + wv;  // 0 .. 32*512-1
  const float* grow = g + (size_t)row * 512;
  float dsum = 0.f;
  for (int i = 0; i < 4; ++i) {
    int o = (lane + i * 32) * 4;
    float4 v = *(const float4*)(grow + o);
    *(float4*)&sG[wv][o] = v;
    dsum += v.x + v.y + v.z + v.w;
  }
  for (int off = 16; off >= 1; off >>= 1) dsum += __shfl_xor(dsum, off);
  __syncthreads();

  const float* erow = e + (size_t)row * 512 * 16;
  int wsub = lane >> 2;  // 0..7
  int f4 = lane & 3;     // float4 chunk of the 16 edge feats
  float4 acc = make_float4(0.f, 0.f, 0.f, 0.f);
  for (int i = 0; i < 64; ++i) {
    int w = i * 8 + wsub;
    float gv = sG[wv][w];
    float4 ev = *(const float4*)(erow + (size_t)w * 16 + f4 * 4);
    acc.x += gv * ev.x; acc.y += gv * ev.y;
    acc.z += gv * ev.z; acc.w += gv * ev.w;
  }
  for (int off = 4; off <= 16; off <<= 1) {
    acc.x += __shfl_xor(acc.x, off); acc.y += __shfl_xor(acc.y, off);
    acc.z += __shfl_xor(acc.z, off); acc.w += __shfl_xor(acc.w, off);
  }
  if (lane < 4) *(float4*)(me + (size_t)row * 16 + lane * 4) = acc;
  if (lane == 0) deg[row] = dsum;
}

// ---------------------------------------------------------------------------
// Kernel 2: one message-passing step.
// Block = (batch b, 128-row slab). Phase A: mh = g @ h via WMMA over K=512.
// Phase B: h' = blend_d [m @ H[d]] with per-row degree one-hot, K=80 pad 96.
// ---------------------------------------------------------------------------
__global__ __launch_bounds__(256) void step_kernel(const float* __restrict__ g,
                                                   const float* __restrict__ h,
                                                   const float* __restrict__ me,
                                                   const float* __restrict__ deg,
                                                   const float* __restrict__ H,
                                                   float* __restrict__ hout) {
  __shared__ __align__(16) _Float16 sA[128 * 32];   // g slab chunk (row-major)
  __shared__ __align__(16) _Float16 sBt[64 * 32];   // h chunk, N-major
  __shared__ __align__(16) _Float16 sM[128 * 96];   // m = [mh | me | pad]
  __shared__ __align__(16) _Float16 sHT[64 * 96];   // H[d]^T, N-major, K pad
  __shared__ float sDeg[128];

  int b = blockIdx.x >> 2;
  int vBase = (blockIdx.x & 3) * 128;
  int tid = threadIdx.x, lane = tid & 31, wv = tid >> 5;
  const float* gB = g + ((size_t)b * 512 + vBase) * 512;
  const float* hB = h + (size_t)b * 512 * 64;

  v8f zero8 = {};
  v8f acc[4];
  for (int nt = 0; nt < 4; ++nt) acc[nt] = zero8;

  int mrow = wv * 16 + (lane & 15);
  for (int kc = 0; kc < 512; kc += 32) {
    for (int i = 0; i < 4; ++i) {          // 128x32 f32 -> f16, float4 loads
      int idx4 = tid + i * 256;
      int r = idx4 >> 3, c4 = idx4 & 7;
      float4 v = *(const float4*)(gB + (size_t)r * 512 + kc + c4 * 4);
      _Float16* d = &sA[r * 32 + c4 * 4];
      d[0] = (_Float16)v.x; d[1] = (_Float16)v.y;
      d[2] = (_Float16)v.z; d[3] = (_Float16)v.w;
    }
    for (int i = 0; i < 8; ++i) {          // h[kc+k][n] -> sBt[n][k]
      int idx = tid + i * 256;
      int k = idx >> 6, n = idx & 63;
      sBt[n * 32 + k] = (_Float16)hB[(size_t)(kc + k) * 64 + n];
    }
    __syncthreads();
    v16h aF = load_frag(sA, mrow, 32, 0, lane);
    for (int nt = 0; nt < 4; ++nt) {
      v16h bF = load_frag(sBt, nt * 16 + (lane & 15), 32, 0, lane);
      acc[nt] = WMMA16(aF, bF, acc[nt]);
    }
    __syncthreads();
  }

  // Stage m = [mh (64) | me (16) | zero pad (16)] into LDS.
  {
    int hp = lane >> 4, nn = lane & 15;
    for (int nt = 0; nt < 4; ++nt)
      for (int r = 0; r < 8; ++r)
        sM[(wv * 16 + r + 8 * hp) * 96 + nt * 16 + nn] = (_Float16)acc[nt][r];
  }
  for (int i = 0; i < 8; ++i) {
    int idx = tid + i * 256;
    int r = idx >> 4, c = idx & 15;
    sM[r * 96 + 64 + c] =
        (_Float16)me[((size_t)b * 512 + vBase + r) * 16 + c];
    sM[r * 96 + 80 + c] = (_Float16)0.f;
  }
  if (tid < 128) sDeg[tid] = deg[(size_t)b * 512 + vBase + tid];

  v8f hacc[4];
  for (int nt = 0; nt < 4; ++nt) hacc[nt] = zero8;

  for (int d = 0; d < 4; ++d) {
    const float* Hd = H + (size_t)d * 80 * 64;
    for (int i = 0; i < 20; ++i) {        // H[d][k][n] -> sHT[n][k], k<80
      int idx = tid + i * 256;
      int k = idx >> 6, n = idx & 63;
      sHT[n * 96 + k] = (_Float16)Hd[(size_t)k * 64 + n];
    }
    for (int i = 0; i < 4; ++i) {         // zero pad k = 80..95
      int idx = tid + i * 256;
      int n = idx >> 4, k = 80 + (idx & 15);
      sHT[n * 96 + k] = (_Float16)0.f;
    }
    __syncthreads();

    v8f a2[4];
    for (int nt = 0; nt < 4; ++nt) a2[nt] = zero8;
    for (int kc = 0; kc < 96; kc += 32) {
      v16h aF = load_frag(sM, mrow, 96, kc, lane);
      for (int nt = 0; nt < 4; ++nt) {
        v16h bF = load_frag(sHT, nt * 16 + (lane & 15), 96, kc, lane);
        a2[nt] = WMMA16(aF, bF, a2[nt]);
      }
    }
    int hp = lane >> 4;
    float dd = (float)(d + 1);
    for (int r = 0; r < 8; ++r) {
      float sel = (sDeg[wv * 16 + r + 8 * hp] == dd) ? 1.f : 0.f;
      for (int nt = 0; nt < 4; ++nt) hacc[nt][r] += sel * a2[nt][r];
    }
    __syncthreads();
  }

  {
    int hp = lane >> 4, nn = lane & 15;
    float* out = hout + ((size_t)b * 512 + vBase) * 64;
    for (int nt = 0; nt < 4; ++nt)
      for (int r = 0; r < 8; ++r)
        out[(size_t)(wv * 16 + r + 8 * hp) * 64 + nt * 16 + nn] = hacc[nt][r];
  }
}

// ---------------------------------------------------------------------------
// Kernel 3: readout for one layer: accG[b] += sum_v mask * softmax(h @ W).
// Block = (b, 128-node slab); wave owns a 16-row strip x 128 cols.
// ---------------------------------------------------------------------------
__global__ __launch_bounds__(256) void readout_kernel(const float* __restrict__ h,
                                                      const float* __restrict__ W,
                                                      float* __restrict__ accG) {
  __shared__ __align__(16) _Float16 sA2[128 * 64];
  __shared__ __align__(16) _Float16 sWT[128 * 64];
  int b = blockIdx.x >> 2;
  int vBase = (blockIdx.x & 3) * 128;
  int tid = threadIdx.x, lane = tid & 31, wv = tid >> 5;

  for (int i = 0; i < 8; ++i) {           // 128x64 slab of h -> f16
    int idx4 = tid + i * 256;             // float4 index over 8192 elems
    int r = idx4 >> 4, c4 = idx4 & 15;
    float4 v = *(const float4*)(h + ((size_t)b * 512 + vBase + r) * 64 + c4 * 4);
    _Float16* d = &sA2[r * 64 + c4 * 4];
    d[0] = (_Float16)v.x; d[1] = (_Float16)v.y;
    d[2] = (_Float16)v.z; d[3] = (_Float16)v.w;
  }
  for (int i = 0; i < 32; ++i) {          // W (64,128) -> sWT[n][k]
    int idx = tid + i * 256;
    int k = idx >> 7, n = idx & 127;
    sWT[n * 64 + k] = (_Float16)W[(size_t)k * 128 + n];
  }
  __syncthreads();

  v8f zero8 = {};
  v8f acc[8];
  for (int nt = 0; nt < 8; ++nt) acc[nt] = zero8;
  int mrow = wv * 16 + (lane & 15);
  for (int kc = 0; kc < 64; kc += 32) {
    v16h aF = load_frag(sA2, mrow, 64, kc, lane);
    for (int nt = 0; nt < 8; ++nt) {
      v16h bF = load_frag(sWT, nt * 16 + (lane & 15), 64, kc, lane);
      acc[nt] = WMMA16(aF, bF, acc[nt]);
    }
  }

  // Per-row masked softmax across the 128 columns (spread over a 16-lane
  // half-wave x 8 accumulators), then sum the strip's 16 rows per column.
  float scale[8];
  for (int r = 0; r < 8; ++r) {
    float mx = acc[0][r], ab = fabsf(acc[0][r]);
    for (int nt = 1; nt < 8; ++nt) {
      mx = fmaxf(mx, acc[nt][r]);
      ab = fmaxf(ab, fabsf(acc[nt][r]));
    }
    for (int off = 1; off <= 8; off <<= 1) {
      mx = fmaxf(mx, __shfl_xor(mx, off));
      ab = fmaxf(ab, __shfl_xor(ab, off));
    }
    float s = 0.f;
    for (int nt = 0; nt < 8; ++nt) {
      float ev = __expf(acc[nt][r] - mx);
      acc[nt][r] = ev;
      s += ev;
    }
    for (int off = 1; off <= 8; off <<= 1) s += __shfl_xor(s, off);
    scale[r] = (ab > 0.f) ? (1.f / s) : 0.f;  // mask all-zero rows
  }
  for (int nt = 0; nt < 8; ++nt) {
    float cs = 0.f;
    for (int r = 0; r < 8; ++r) cs += acc[nt][r] * scale[r];
    cs += __shfl_xor(cs, 16);  // combine the two 8-row lane halves
    if (lane < 16) atomicAdd(&accG[b * 128 + nt * 16 + lane], cs);
  }
}

// ---------------------------------------------------------------------------
// Kernel 4: out[b] = softmax(accG[b] @ W_final + b_final)
// ---------------------------------------------------------------------------
__global__ __launch_bounds__(32) void final_kernel(const float* __restrict__ accG,
                                                   const float* __restrict__ Wf,
                                                   const float* __restrict__ bf,
                                                   float* __restrict__ out) {
  int b = blockIdx.x;
  int t = threadIdx.x & 15;
  float o = bf[t];
  for (int j = 0; j < 128; ++j) o += accG[b * 128 + j] * Wf[j * 16 + t];
  float mx = o;
  for (int off = 1; off <= 8; off <<= 1) mx = fmaxf(mx, __shfl_xor(mx, off));
  float ev = __expf(o - mx);
  float s = ev;
  for (int off = 1; off <= 8; off <<= 1) s += __shfl_xor(s, off);
  if (threadIdx.x < 16) out[b * 16 + t] = ev / s;
}

__global__ void zero_kernel(float* p, int n) {
  int i = blockIdx.x * blockDim.x + threadIdx.x;
  if (i < n) p[i] = 0.f;
}

extern "C" void kernel_launch(void* const* d_in, const int* in_sizes, int n_in,
                              void* d_out, int out_size, void* d_ws, size_t ws_size,
                              hipStream_t stream) {
  const float* g      = (const float*)d_in[0];
  const float* h_in   = (const float*)d_in[1];
  const float* e      = (const float*)d_in[2];
  const float* H0     = (const float*)d_in[3];
  const float* H1     = (const float*)d_in[4];
  const float* H2     = (const float*)d_in[5];
  const float* W0     = (const float*)d_in[6];
  const float* W1     = (const float*)d_in[7];
  const float* W2     = (const float*)d_in[8];
  const float* W3     = (const float*)d_in[9];
  const float* Wf     = (const float*)d_in[10];
  const float* bf     = (const float*)d_in[11];
  float* out = (float*)d_out;

  // Workspace layout (floats): me | deg | h1 | h2 | h3 | accG  (~13.7 MB)
  float* me   = (float*)d_ws;
  float* deg  = me + 32 * 512 * 16;
  float* h1   = deg + 32 * 512;
  float* h2   = h1 + 32 * 512 * 64;
  float* h3   = h2 + 32 * 512 * 64;
  float* accG = h3 + 32 * 512 * 64;

  me_deg_kernel<<<2048, 256, 0, stream>>>(g, e, me, deg);
  zero_kernel<<<16, 256, 0, stream>>>(accG, 32 * 128);

  step_kernel<<<128, 256, 0, stream>>>(g, h_in, me, deg, H0, h1);
  step_kernel<<<128, 256, 0, stream>>>(g, h1, me, deg, H1, h2);
  step_kernel<<<128, 256, 0, stream>>>(g, h2, me, deg, H2, h3);

  readout_kernel<<<128, 256, 0, stream>>>(h_in, W0, accG);
  readout_kernel<<<128, 256, 0, stream>>>(h1, W1, accG);
  readout_kernel<<<128, 256, 0, stream>>>(h2, W2, accG);
  readout_kernel<<<128, 256, 0, stream>>>(h3, W3, accG);

  final_kernel<<<32, 32, 0, stream>>>(accG, Wf, bf, out);
}
